// ImageClassifier_12687333392910
// MI455X (gfx1250) — compile-verified, matrix operand order checked by reading, not run on hardware
//
#include <hip/hip_runtime.h>
#include <hip/hip_bf16.h>

typedef __attribute__((ext_vector_type(16))) _Float16 v16h;
typedef __attribute__((ext_vector_type(8)))  _Float16 v8h;
typedef __attribute__((ext_vector_type(8)))  float    v8f;

#define ODE_NWG 49

// Dormand-Prince A coefficients (stage s uses k_0..k_{s-1})
__device__ __constant__ float c_AT[6][5] = {
  {0.f, 0.f, 0.f, 0.f, 0.f},
  {1.f/5.f, 0.f, 0.f, 0.f, 0.f},
  {3.f/40.f, 9.f/40.f, 0.f, 0.f, 0.f},
  {44.f/45.f, -56.f/15.f, 32.f/9.f, 0.f, 0.f},
  {19372.f/6561.f, -25360.f/2187.f, 64448.f/6561.f, -212.f/729.f, 0.f},
  {9017.f/3168.f, -355.f/33.f, 46732.f/5247.f, 49.f/176.f, -5103.f/18656.f}
};

__device__ __forceinline__ v8h v8h_zero(){ v8h r; __builtin_memset(&r, 0, sizeof(r)); return r; }

// ---- WMMA fragment load -----------------------------------------------------
// 16-bit A (16x32 MxK): lane L holds row m=(L&15); VGPRs 0..3 = K 8*half+{0..7},
// VGPRs 4..7 = K 8*half+16+{0..7} (half=L>>4) -> two contiguous 16B loads/lane.
// B (32x16 KxN) mirrors with N in place of M, so Bt[N][K] row-major reuses this.
__device__ __forceinline__ v16h ldfrag(const _Float16* base, int ld, int row, int k0, int lane){
  const int hf = (lane >> 4) & 1;
  const _Float16* p = base + (size_t)(row + (lane & 15)) * ld + k0 + hf * 8;
  union { v16h v; v8h h[2]; } t;
  t.h[0] = *(const v8h*)p;
  t.h[1] = *(const v8h*)(p + 16);
  return t.v;
}

__device__ __forceinline__ v8f wmma16(v16h a, v16h b, v8f c){
  return __builtin_amdgcn_wmma_f32_16x16x32_f16(false, a, false, b, (short)0, c, false, false);
}

// ---- software grid barrier (monotonic counter; reset by hipMemsetAsync) -----
__device__ __forceinline__ void grid_sync(unsigned* ctrl, unsigned target){
  __threadfence();
  __syncthreads();
  if (threadIdx.x == 0){
    atomicAdd(ctrl, 1u);
    volatile unsigned* vp = ctrl;
    while (*vp < target) __builtin_amdgcn_s_sleep(1);
  }
  __syncthreads();
  __threadfence();
}

// ---- weight prep: dst[n*K+k] = pad(src[k][n]) as f16 ------------------------
__global__ void k_transpose_pad_f16(_Float16* __restrict__ dst, const float* __restrict__ src,
                                    int N, int K, int Nsrc, int Ksrc){
  int idx = blockIdx.x * 256 + threadIdx.x;
  if (idx >= N * K) return;
  int n = idx / K, k = idx % K;
  float v = (n < Nsrc && k < Ksrc) ? src[(size_t)k * Nsrc + n] : 0.f;
  dst[idx] = (_Float16)v;
}

// ---- conv weight flatten: OIHW [64][64][4][4] f32 -> [n=64][k=(ky*4+kx)*64+ci] f16
__global__ void k_prep_convw(_Float16* __restrict__ dst, const float* __restrict__ w){
  int idx = blockIdx.x * 256 + threadIdx.x;      // 65536
  int n = idx >> 10, k = idx & 1023;
  int wi = k >> 6, ci = k & 63, ky = wi >> 2, kx = wi & 3;
  dst[idx] = (_Float16)w[((n * 64 + ci) * 4 + ky) * 4 + kx];
}

// ---- conv1: 3x3 s1 p0, NCHW f32 in -> NHWC f16 out [128,30,30,64], ReLU -----
__global__ void k_conv1(const float* __restrict__ x, const float* __restrict__ w,
                        const float* __restrict__ b, _Float16* __restrict__ out){
  int idx = blockIdx.x * 256 + threadIdx.x;            // 128*30*30*64
  int c = idx & 63; int t = idx >> 6;
  int ox = t % 30; t /= 30; int oy = t % 30; int bb = t / 30;
  float acc = b[c];
  const float* xb = x + (size_t)bb * 3 * 1024;
  const float* wc = w + c * 27;
  #pragma unroll
  for (int ci = 0; ci < 3; ++ci)
    #pragma unroll
    for (int ky = 0; ky < 3; ++ky)
      #pragma unroll
      for (int kx = 0; kx < 3; ++kx)
        acc += xb[ci * 1024 + (oy + ky) * 32 + (ox + kx)] * wc[(ci * 3 + ky) * 3 + kx];
  out[idx] = (_Float16)fmaxf(acc, 0.f);
}

// ---- conv2 implicit-GEMM WMMA: M=28800 pixels, N=64, K=1024 -----------------
__global__ void __launch_bounds__(256, 1)
k_conv2_wmma(const _Float16* __restrict__ hin, const _Float16* __restrict__ wt,
             const float* __restrict__ bias, _Float16* __restrict__ out){
  __shared__ __align__(16) _Float16 At[128 * 128];
  const int tid = threadIdx.x, lane = tid & 31, wave = tid >> 5;
  const int m0 = (wave >> 1) * 32, n0 = (wave & 1) * 32;
  const int cl = lane & 15, hf = (lane >> 4) & 1;
  const int pix0 = blockIdx.x * 128;

  v8f acc[4] = {};
  for (int ch = 0; ch < 8; ++ch){
    __syncthreads();
    for (int it = tid; it < 2048; it += 256){
      const int p = it >> 4, r = it & 15;
      const int wl = r >> 3, seg = r & 7;
      const int gp = pix0 + p;
      const int b = gp / 225, q = gp % 225;
      const int oy = q / 15, ox = q % 15;
      const int wi = ch * 2 + wl;
      const int ky = wi >> 2, kx = wi & 3;
      const int iy = 2 * oy - 1 + ky, ix = 2 * ox - 1 + kx;
      v8h val = v8h_zero();
      if ((unsigned)iy < 30u && (unsigned)ix < 30u)
        val = *(const v8h*)(hin + (size_t)((b * 30 + iy) * 30 + ix) * 64 + seg * 8);
      *(v8h*)(At + p * 128 + wl * 64 + seg * 8) = val;
    }
    __syncthreads();
    #pragma unroll
    for (int kk = 0; kk < 128; kk += 32){
      v16h a0  = ldfrag(At, 128, m0,      kk, lane);
      v16h a1  = ldfrag(At, 128, m0 + 16, kk, lane);
      v16h bb0 = ldfrag(wt, 1024, n0,      ch * 128 + kk, lane);
      v16h bb1 = ldfrag(wt, 1024, n0 + 16, ch * 128 + kk, lane);
      acc[0] = wmma16(a0, bb0, acc[0]);
      acc[1] = wmma16(a0, bb1, acc[1]);
      acc[2] = wmma16(a1, bb0, acc[2]);
      acc[3] = wmma16(a1, bb1, acc[3]);
    }
  }
  #pragma unroll
  for (int t2 = 0; t2 < 4; ++t2){
    union { v8f v; float f[8]; } ua; ua.v = acc[t2];
    const int mm = m0 + (t2 >> 1) * 16 + hf * 8;
    const int nn = n0 + (t2 & 1) * 16 + cl;
    const float bv = bias[nn];
    #pragma unroll
    for (int r = 0; r < 8; ++r)
      out[(size_t)(pix0 + mm + r) * 64 + nn] = (_Float16)fmaxf(ua.f[r] + bv, 0.f);
  }
}

// ---- conv3 implicit-GEMM WMMA: M=6272 pixels, N=64, K=1024 -> y f32 ---------
__global__ void __launch_bounds__(256, 1)
k_conv3_wmma(const _Float16* __restrict__ hin, const _Float16* __restrict__ wt,
             const float* __restrict__ bias, float* __restrict__ y){
  __shared__ __align__(16) _Float16 At[128 * 128];
  const int tid = threadIdx.x, lane = tid & 31, wave = tid >> 5;
  const int m0 = (wave >> 1) * 32, n0 = (wave & 1) * 32;
  const int cl = lane & 15, hf = (lane >> 4) & 1;
  const int pix0 = blockIdx.x * 128;

  v8f acc[4] = {};
  for (int ch = 0; ch < 8; ++ch){
    __syncthreads();
    for (int it = tid; it < 2048; it += 256){
      const int p = it >> 4, r = it & 15;
      const int wl = r >> 3, seg = r & 7;
      const int gp = pix0 + p;
      const int b = gp / 49, q = gp % 49;
      const int oy = q / 7, ox = q % 7;
      const int wi = ch * 2 + wl;
      const int ky = wi >> 2, kx = wi & 3;
      const int iy = 2 * oy - 1 + ky, ix = 2 * ox - 1 + kx;
      v8h val = v8h_zero();
      if ((unsigned)iy < 15u && (unsigned)ix < 15u)
        val = *(const v8h*)(hin + (size_t)((b * 15 + iy) * 15 + ix) * 64 + seg * 8);
      *(v8h*)(At + p * 128 + wl * 64 + seg * 8) = val;
    }
    __syncthreads();
    #pragma unroll
    for (int kk = 0; kk < 128; kk += 32){
      v16h a0  = ldfrag(At, 128, m0,      kk, lane);
      v16h a1  = ldfrag(At, 128, m0 + 16, kk, lane);
      v16h bb0 = ldfrag(wt, 1024, n0,      ch * 128 + kk, lane);
      v16h bb1 = ldfrag(wt, 1024, n0 + 16, ch * 128 + kk, lane);
      acc[0] = wmma16(a0, bb0, acc[0]);
      acc[1] = wmma16(a0, bb1, acc[1]);
      acc[2] = wmma16(a1, bb0, acc[2]);
      acc[3] = wmma16(a1, bb1, acc[3]);
    }
  }
  // scatter into y in NCHW-flat order: y[b*3136 + c*49 + (oy*7+ox)]
  #pragma unroll
  for (int t2 = 0; t2 < 4; ++t2){
    union { v8f v; float f[8]; } ua; ua.v = acc[t2];
    const int mm = m0 + (t2 >> 1) * 16 + hf * 8;
    const int nn = n0 + (t2 & 1) * 16 + cl;
    const float bv = bias[nn];
    #pragma unroll
    for (int r = 0; r < 8; ++r){
      const int gp = pix0 + mm + r;
      const int b = gp / 49, pp = gp % 49;
      y[(size_t)b * 3136 + nn * 49 + pp] = ua.f[r] + bv;
    }
  }
}

// ---- persistent DOPRI5 integrator: 49 WGs x 256 threads ---------------------
// Dynamic LDS (272KB): full per-WG ODE state resident in the WGP's 320KB LDS.
//   yhL 16K | zh 16K | w1L 8K | w2L 8K | kL 6x32K=192K | yL 32K
__global__ void __launch_bounds__(256, 1)
k_ode(const float* __restrict__ b1, const float* __restrict__ b2,
      const _Float16* __restrict__ w1t, const _Float16* __restrict__ w2t,
      float* __restrict__ y, _Float16* __restrict__ yh,
      float* __restrict__ u, unsigned* ctrl)
{
  extern __shared__ __align__(16) char smem[];
  _Float16* yhL = (_Float16*)(smem);
  _Float16* zh  = (_Float16*)(smem + 16384);
  _Float16* w1L = (_Float16*)(smem + 32768);
  _Float16* w2L = (_Float16*)(smem + 40960);
  float*    kL  = (float*)(smem + 49152);      // [6][8192]
  float*    yL  = (float*)(smem + 245760);     // [8192]

  const int tid  = threadIdx.x;
  const int lane = tid & 31;
  const int wave = tid >> 5;
  const int blk  = blockIdx.x;
  const int c0 = blk * 64;               // this WG's 64-col slice of 3136
  const int m0 = (wave >> 1) * 32;       // 4x2 wave grid tiles 128x64
  const int n0 = (wave & 1) * 32;
  const int cl = lane & 15;
  const int hf = (lane >> 4) & 1;
  const float h = 1.0f / 40.0f;
  const int zbase = blk * 168;           // this WG's zero-slice of u (8192 total)
  unsigned bar = 0;

  // one-time staging: weight slices, y slice, zero both u buffers
  for (int i = tid; i < 512; i += 256){
    const int row = i >> 3, seg = i & 7;
    *(v8h*)(w1L + row * 64 + seg * 8) = *(const v8h*)(w1t + (size_t)row * 3136 + c0 + seg * 8);
    *(v8h*)(w2L + row * 64 + seg * 8) = *(const v8h*)(w2t + (size_t)(c0 + row) * 64 + seg * 8);
  }
  for (int i = tid; i < 8192; i += 256)
    yL[i] = y[(i >> 6) * 3136 + c0 + (i & 63)];
  for (int i = tid; i < 168; i += 256){
    const int j = zbase + i;
    if (j < 8192){ u[j] = 0.f; u[8192 + j] = 0.f; }
  }
  grid_sync(ctrl, bar += ODE_NWG);

  for (int step = 0; step < 40; ++step){
    for (int s = 0; s < 6; ++s){
      float* ucur = u + (s & 1) * 8192;
      float* unxt = u + ((s + 1) & 1) * 8192;

      // Phase A (WG-local): stage input in LDS
      for (int i = tid; i < 8192; i += 256){
        float v = yL[i];
        for (int j = 0; j < s; ++j)
          v += h * c_AT[s][j] * kL[j * 8192 + i];
        yhL[i] = (_Float16)v;
      }
      __syncthreads();

      // Phase B: GEMM1 K-slice partial entirely from LDS -> atomics into ucur
      {
        v8f acc[4] = {};
        #pragma unroll
        for (int kk = 0; kk < 64; kk += 32){
          v16h a0  = ldfrag(yhL, 64, m0,      kk, lane);
          v16h a1  = ldfrag(yhL, 64, m0 + 16, kk, lane);
          v16h bb0 = ldfrag(w1L, 64, n0,      kk, lane);
          v16h bb1 = ldfrag(w1L, 64, n0 + 16, kk, lane);
          acc[0] = wmma16(a0, bb0, acc[0]);
          acc[1] = wmma16(a0, bb1, acc[1]);
          acc[2] = wmma16(a1, bb0, acc[2]);
          acc[3] = wmma16(a1, bb1, acc[3]);
        }
        #pragma unroll
        for (int t2 = 0; t2 < 4; ++t2){
          union { v8f v; float f[8]; } ua; ua.v = acc[t2];
          const int mm = m0 + (t2 >> 1) * 16 + hf * 8;
          const int nn = n0 + (t2 & 1) * 16 + cl;
          #pragma unroll
          for (int r = 0; r < 8; ++r)
            atomicAdd(&ucur[(mm + r) * 64 + nn], ua.f[r]);
        }
      }
      grid_sync(ctrl, bar += ODE_NWG);

      // Phase C: z = relu(u+b1) to LDS; zero our slice of the other u buffer;
      //          GEMM2 -> kL[s] (LDS) + b2; s==5: update yL
      for (int i = tid; i < 8192; i += 256){
        const int cc = i & 63;
        float v = (cc < 50) ? fmaxf(ucur[i] + b1[cc], 0.f) : 0.f;
        zh[i] = (_Float16)v;
      }
      for (int i = tid; i < 168; i += 256){
        const int j = zbase + i;
        if (j < 8192) unxt[j] = 0.f;
      }
      __syncthreads();
      {
        v8f acc[4] = {};
        #pragma unroll
        for (int k0 = 0; k0 < 64; k0 += 32){
          v16h a0  = ldfrag(zh,  64, m0,      k0, lane);
          v16h a1  = ldfrag(zh,  64, m0 + 16, k0, lane);
          v16h bb0 = ldfrag(w2L, 64, n0,      k0, lane);
          v16h bb1 = ldfrag(w2L, 64, n0 + 16, k0, lane);
          acc[0] = wmma16(a0, bb0, acc[0]);
          acc[1] = wmma16(a0, bb1, acc[1]);
          acc[2] = wmma16(a1, bb0, acc[2]);
          acc[3] = wmma16(a1, bb1, acc[3]);
        }
        float* kd = kL + s * 8192;
        #pragma unroll
        for (int t2 = 0; t2 < 4; ++t2){
          union { v8f v; float f[8]; } ua; ua.v = acc[t2];
          const int mm = m0 + (t2 >> 1) * 16 + hf * 8;
          const int nnLoc = n0 + (t2 & 1) * 16 + cl;
          const float bias = b2[c0 + nnLoc];
          #pragma unroll
          for (int r = 0; r < 8; ++r)
            kd[(mm + r) * 64 + nnLoc] = ua.f[r] + bias;
        }
      }
      if (s == 5){
        __syncthreads();
        for (int i = tid; i < 8192; i += 256){
          yL[i] += h * ((35.f/384.f)   * kL[i]            + (500.f/1113.f)   * kL[2*8192 + i]
                      + (125.f/192.f)  * kL[3*8192 + i]   + (-2187.f/6784.f) * kL[4*8192 + i]
                      + (11.f/84.f)    * kL[5*8192 + i]);
        }
      }
      grid_sync(ctrl, bar += ODE_NWG);
    }
  }
  // final y -> f16 global for up1 GEMM
  for (int i = tid; i < 8192; i += 256)
    yh[(i >> 6) * 3136 + c0 + (i & 63)] = (_Float16)yL[i];
}

// ---- up1: [128,3136] @ [3136,512] + bias, ReLU, f32 out ---------------------
__global__ void __launch_bounds__(256)
k_up1(const _Float16* __restrict__ A, const _Float16* __restrict__ Bt,
      const float* __restrict__ bias, float* __restrict__ C){
  const int tid = threadIdx.x, lane = tid & 31, wave = tid >> 5;
  const int m0 = (wave >> 1) * 32;
  const int n0 = blockIdx.x * 64 + (wave & 1) * 32;
  const int cl = lane & 15, hf = (lane >> 4) & 1;
  v8f acc[4] = {};
  for (int k0 = 0; k0 < 3136; k0 += 32){
    v16h a0  = ldfrag(A,  3136, m0,      k0, lane);
    v16h a1  = ldfrag(A,  3136, m0 + 16, k0, lane);
    v16h bb0 = ldfrag(Bt, 3136, n0,      k0, lane);
    v16h bb1 = ldfrag(Bt, 3136, n0 + 16, k0, lane);
    acc[0] = wmma16(a0, bb0, acc[0]);
    acc[1] = wmma16(a0, bb1, acc[1]);
    acc[2] = wmma16(a1, bb0, acc[2]);
    acc[3] = wmma16(a1, bb1, acc[3]);
  }
  #pragma unroll
  for (int t2 = 0; t2 < 4; ++t2){
    union { v8f v; float f[8]; } ua; ua.v = acc[t2];
    const int mm = m0 + (t2 >> 1) * 16 + hf * 8;
    const int nn = n0 + (t2 & 1) * 16 + cl;
    const float bv = bias[nn];
    #pragma unroll
    for (int r = 0; r < 8; ++r)
      C[(mm + r) * 512 + nn] = fmaxf(ua.f[r] + bv, 0.f);
  }
}

// ---- up2: [128,512] @ [512,10] + bias -> d_out ------------------------------
__global__ void k_up2(const float* __restrict__ hin, const float* __restrict__ w,
                      const float* __restrict__ b, float* __restrict__ out){
  int idx = blockIdx.x * 256 + threadIdx.x;
  if (idx >= 1280) return;
  int bb = idx / 10, j = idx % 10;
  float acc = b[j];
  const float* hp = hin + bb * 512;
  for (int kk = 0; kk < 512; ++kk) acc += hp[kk] * w[kk * 10 + j];
  out[idx] = acc;
}

extern "C" void kernel_launch(void* const* d_in, const int* in_sizes, int n_in,
                              void* d_out, int out_size, void* d_ws, size_t ws_size,
                              hipStream_t stream){
  (void)in_sizes; (void)n_in; (void)out_size; (void)ws_size;
  const float* x   = (const float*)d_in[0];
  const float* c1w = (const float*)d_in[1];
  const float* c1b = (const float*)d_in[2];
  const float* c2w = (const float*)d_in[3];
  const float* c2b = (const float*)d_in[4];
  const float* c3w = (const float*)d_in[5];
  const float* c3b = (const float*)d_in[6];
  const float* ow1 = (const float*)d_in[7];
  const float* ob1 = (const float*)d_in[8];
  const float* ow2 = (const float*)d_in[9];
  const float* ob2 = (const float*)d_in[10];
  const float* u1w = (const float*)d_in[11];
  const float* u1b = (const float*)d_in[12];
  const float* u2w = (const float*)d_in[13];
  const float* u2b = (const float*)d_in[14];
  float* out = (float*)d_out;

  char* ws = (char*)d_ws;
  size_t off = 0;
  auto alloc = [&](size_t bytes) -> char* {
    char* p = ws + off; off = (off + bytes + 255) & ~(size_t)255; return p;
  };
  unsigned*  ctrl = (unsigned*)alloc(256);
  _Float16*  yh   = (_Float16*)alloc((size_t)128 * 3136 * 2);
  _Float16*  w1t  = (_Float16*)alloc((size_t)64 * 3136 * 2);
  _Float16*  w2t  = (_Float16*)alloc((size_t)3136 * 64 * 2);
  _Float16*  u1t  = (_Float16*)alloc((size_t)512 * 3136 * 2);
  _Float16*  w2ct = (_Float16*)alloc((size_t)64 * 1024 * 2);
  _Float16*  w3ct = (_Float16*)alloc((size_t)64 * 1024 * 2);
  float*     u    = (float*)alloc((size_t)2 * 128 * 64 * 4);
  float*     y    = (float*)alloc((size_t)128 * 3136 * 4);
  _Float16*  h1   = (_Float16*)alloc((size_t)128 * 30 * 30 * 64 * 2);
  _Float16*  h2   = (_Float16*)alloc((size_t)128 * 15 * 15 * 64 * 2);
  float*     h512 = (float*)alloc((size_t)128 * 512 * 4);

  hipMemsetAsync(ctrl, 0, 256, stream);
  k_transpose_pad_f16<<<784,  256, 0, stream>>>(w1t, ow1, 64,   3136, 50,   3136);
  k_transpose_pad_f16<<<784,  256, 0, stream>>>(w2t, ow2, 3136, 64,   3136, 50);
  k_transpose_pad_f16<<<6272, 256, 0, stream>>>(u1t, u1w, 512,  3136, 512,  3136);
  k_prep_convw<<<256, 256, 0, stream>>>(w2ct, c2w);
  k_prep_convw<<<256, 256, 0, stream>>>(w3ct, c3w);
  k_conv1<<<28800, 256, 0, stream>>>(x, c1w, c1b, h1);
  k_conv2_wmma<<<225, 256, 0, stream>>>(h1, w2ct, c2b, h2);
  k_conv3_wmma<<<49,  256, 0, stream>>>(h2, w3ct, c3b, y);

  const int odeLds = 278528;  // 272KB dynamic LDS (<= 320KB WGP LDS)
  hipFuncSetAttribute((const void*)k_ode, hipFuncAttributeMaxDynamicSharedMemorySize, odeLds);
  k_ode<<<ODE_NWG, 256, odeLds, stream>>>(ob1, ob2, w1t, w2t, y, yh, u, ctrl);

  k_up1<<<8, 256, 0, stream>>>(yh, u1t, u1b, h512);
  k_up2<<<5, 256, 0, stream>>>(h512, u2w, u2b, out);
}
